// Bert_71554155151426
// MI455X (gfx1250) — matrix operand order repaired and checked by reading, not compiled, not run on hardware
//
#include <hip/hip_runtime.h>
#include <hip/hip_fp16.h>

typedef __attribute__((ext_vector_type(16))) _Float16 v16h;
typedef __attribute__((ext_vector_type(8)))  float    v8f;
typedef __attribute__((ext_vector_type(4)))  float    f32x4;
typedef __attribute__((ext_vector_type(4)))  int      i32x4;

#define B_  256
#define L_  100
#define H_  384
#define NH_ 12
#define DH_ 32
#define NL_ 6
#define FF_ 1536
#define M_  (B_ * L_)   // 25600
#define LP_ 112         // L padded to 7*16
#define LK_ 128         // K dim of P@V padded to 4*32

union V16U { v16h v; f32x4 f[2]; };

// ---- CDNA5 async global->LDS copy (ASYNCcnt-tracked), guarded by __has_builtin
#if defined(__has_builtin)
#if __has_builtin(__builtin_amdgcn_global_load_async_to_lds_b128)
#define HAVE_ASYNC_LDS 1
#endif
#if __has_builtin(__builtin_amdgcn_s_wait_asynccnt)
#define HAVE_WAIT_ASYNC 1
#endif
#endif

typedef __attribute__((address_space(1))) i32x4 as1_i32x4;  // global int4*
typedef __attribute__((address_space(3))) i32x4 as3_i32x4;  // LDS int4*

__device__ __forceinline__ void wait_async_zero() {
#if defined(HAVE_ASYNC_LDS)
#if defined(HAVE_WAIT_ASYNC)
  __builtin_amdgcn_s_wait_asynccnt(0);
#else
  asm volatile("s_wait_asynccnt 0x0" ::: "memory");
#endif
#endif
}

// 16B per-lane async copy global->LDS; completion via ASYNCcnt.
__device__ __forceinline__ void async_copy_b128(const void* g, void* l) {
#if defined(HAVE_ASYNC_LDS)
  __builtin_amdgcn_global_load_async_to_lds_b128(
      (as1_i32x4*)(unsigned long long)g,  // same bit pattern: flat==global addr
      (as3_i32x4*)l,                      // addrspacecast generic->LDS
      0, 0);
#else
  *(f32x4*)l = *(const f32x4*)g;
#endif
}

__device__ __forceinline__ v8f wmma16(v16h a, v16h b, v8f c) {
  return __builtin_amdgcn_wmma_f32_16x16x32_f16(false, a, false, b, (short)0, c,
                                                false, false);
}

// ---------------------------------------------------------------- cast f32->f16
__global__ __launch_bounds__(256) void cast_kernel(const float* __restrict__ s,
                                                   _Float16* __restrict__ d,
                                                   size_t n) {
  size_t stride = (size_t)gridDim.x * blockDim.x;
  for (size_t i = (size_t)blockIdx.x * blockDim.x + threadIdx.x; i < n; i += stride)
    d[i] = (_Float16)s[i];
}

// ---------------------------------------------------------------- embeddings
__global__ __launch_bounds__(128) void embed_kernel(
    const int* __restrict__ x, const int* __restrict__ seg,
    const float* __restrict__ tok, const float* __restrict__ se,
    const float* __restrict__ pe, float* __restrict__ acc,
    _Float16* __restrict__ acch) {
  int row = blockIdx.x;               // 0..B*L-1
  int l = row % L_;
  int xv = x[row], sv = seg[row];
  float np = (xv != 0) ? 1.f : 0.f;
  for (int f = threadIdx.x; f < H_; f += 128) {
    float h = (tok[(size_t)xv * H_ + f] + se[(size_t)sv * H_ + f] +
               pe[(size_t)l * H_ + f]) * np;
    acc[(size_t)row * H_ + f] = h;
    acch[(size_t)row * H_ + f] = (_Float16)h;
  }
}

// ---------------------------------------------------------------- dense GEMM
// C[M,N] = A[M,K](f16) @ W[K,N](f16) + bias[N], fp32 out.
// Block: 256 thr = 8 waves, tile 128x128, BK=32. Wave: 32x64 (2x4 wmma frags).
// A tile staged via GLOBAL_LOAD_ASYNC_TO_LDS_B128 when available.
__global__ __launch_bounds__(256) void gemm_f16(
    const _Float16* __restrict__ A, const _Float16* __restrict__ W,
    const float* __restrict__ bias, float* __restrict__ C, int N, int K) {
  __shared__ _Float16 lsA[128][40];   // [m][k]
  __shared__ _Float16 lsB[128][40];   // [n][k] (transposed)
  const int bm = blockIdx.y * 128;
  const int bn = blockIdx.x * 128;
  const int tid = threadIdx.x;
  const int lane = tid & 31, wave = tid >> 5;
  const int wm0 = (wave >> 1) * 32, wn0 = (wave & 1) * 64;

  v8f zero = {0.f, 0.f, 0.f, 0.f, 0.f, 0.f, 0.f, 0.f};
  v8f accv[2][4];
#pragma unroll
  for (int i = 0; i < 2; ++i)
#pragma unroll
    for (int j = 0; j < 4; ++j) accv[i][j] = zero;

  const int ra = tid >> 1;            // A stage: row 0..127
  const int ca = (tid & 1) * 16;      // halves ca..ca+15
  const int kkb = tid >> 3;           // B stage: k-row 0..31
  const int nnb = (tid & 7) * 16;     // 16 consecutive n

  for (int k0 = 0; k0 < K; k0 += 32) {
    __syncthreads();
    // stage A (row major): per-lane 2 x 16B  global -> LDS (async on CDNA5)
    {
      const _Float16* gA = A + (size_t)(bm + ra) * K + k0 + ca;
      async_copy_b128(gA, &lsA[ra][ca]);
      async_copy_b128(gA + 8, &lsA[ra][ca + 8]);
      if (k0 + 32 < K)
        __builtin_prefetch(A + (size_t)(bm + ra) * K + k0 + 32 + ca, 0, 0);
    }
    // stage B transposed (needs data shuffle -> manual path)
    {
      union { f32x4 f[2]; _Float16 h[16]; } tb;
      const f32x4* g = (const f32x4*)(W + (size_t)(k0 + kkb) * N + bn + nnb);
      tb.f[0] = g[0]; tb.f[1] = g[1];
      if (k0 + 32 < K)
        __builtin_prefetch(W + (size_t)(k0 + 32 + kkb) * N + bn + nnb, 0, 0);
#pragma unroll
      for (int i = 0; i < 16; ++i) lsB[nnb + i][kkb] = tb.h[i];
    }
    wait_async_zero();   // own wave's async copies landed
    __syncthreads();     // everyone's copies visible

    const int mr = lane & 15;
    const int kbA = (lane >> 4) * 8;    // A: half-wave K interleave {kb, 16+kb}
    const int kbB = (lane >> 4) * 16;   // unused
    V16U af[2], bf[4];
#pragma unroll
    for (int i = 0; i < 2; ++i) {
      const _Float16* p = &lsA[wm0 + i * 16 + mr][kbA];
      af[i].f[0] = *(const f32x4*)p;
      af[i].f[1] = *(const f32x4*)(p + 16);
    }
#pragma unroll
    for (int j = 0; j < 4; ++j) {
      const _Float16* p = &lsB[wn0 + j * 16 + mr][kbA];
      bf[j].f[0] = *(const f32x4*)p;
      bf[j].f[1] = *(const f32x4*)(p + 16);
    }
#pragma unroll
    for (int i = 0; i < 2; ++i)
#pragma unroll
      for (int j = 0; j < 4; ++j)
        accv[i][j] = wmma16(af[i].v, bf[j].v, accv[i][j]);
  }

  // epilogue: C layout — col = lane&15, row = vgpr + 8*(lane>=16)
  const int nc = lane & 15;
  const int mo = (lane >> 4) * 8;
#pragma unroll
  for (int i = 0; i < 2; ++i)
#pragma unroll
    for (int j = 0; j < 4; ++j) {
      int col = bn + wn0 + j * 16 + nc;
      float bb = bias ? bias[col] : 0.f;
#pragma unroll
      for (int r = 0; r < 8; ++r) {
        int row = bm + wm0 + i * 16 + mo + r;
        C[(size_t)row * N + col] = accv[i][j][r] + bb;
      }
    }
}

// ---------------------------------------------------------------- attention
// One block per (batch b, head hn). q/k/v fp32 in interleaved-head layout
// (feature = d*NH + head); output f16 in concat layout (feature = head*DH + d).
__global__ __launch_bounds__(256) void attn_kernel(
    const float* __restrict__ q, const float* __restrict__ k,
    const float* __restrict__ v, const int* __restrict__ x,
    _Float16* __restrict__ aout) {
  __shared__ _Float16 lq[LP_][40];      // [l][d]
  __shared__ _Float16 lk[LP_][40];      // [m][d]
  __shared__ _Float16 lvT[DH_][136];    // [d][m], m padded to 128
  __shared__ float    ls[LP_][116];     // scores fp32; probs f16 overlaid per-row

  const int b = blockIdx.x, hn = blockIdx.y;
  const int tid = threadIdx.x, lane = tid & 31, wave = tid >> 5;

  for (int idx = tid; idx < LP_ * DH_; idx += 256) {
    int l = idx >> 5, d = idx & 31;
    float qv = 0.f, kv = 0.f, vv = 0.f;
    if (l < L_) {
      size_t base = ((size_t)b * L_ + l) * H_ + d * NH_ + hn;
      qv = q[base]; kv = k[base]; vv = v[base];
    }
    lq[l][d] = (_Float16)qv;
    lk[l][d] = (_Float16)kv;
    lvT[d][l] = (_Float16)vv;
  }
  for (int idx = tid; idx < DH_ * (LK_ - LP_); idx += 256) {
    int d = idx >> 4, m = LP_ + (idx & 15);
    lvT[d][m] = (_Float16)0.f;
  }
  __syncthreads();

  const int mr = lane & 15;
  const int kbA = (lane >> 4) * 8;
  const int kbB = (lane >> 4) * 16;
  const int mo = (lane >> 4) * 8;
  v8f zero = {0.f, 0.f, 0.f, 0.f, 0.f, 0.f, 0.f, 0.f};

  // S = Q K^T : 49 tiles, DH=32 -> one wmma each
  for (int t = wave; t < 49; t += 8) {
    int tm = t / 7, tn = t % 7;
    V16U af, bf;
    const _Float16* pa = &lq[tm * 16 + mr][kbA];
    af.f[0] = *(const f32x4*)pa; af.f[1] = *(const f32x4*)(pa + 16);
    const _Float16* pb = &lk[tn * 16 + mr][kbA];
    bf.f[0] = *(const f32x4*)pb; bf.f[1] = *(const f32x4*)(pb + 16);
    v8f c = wmma16(af.v, bf.v, zero);
#pragma unroll
    for (int r = 0; r < 8; ++r)
      ls[tm * 16 + mo + r][tn * 16 + (lane & 15)] = c[r];
  }
  __syncthreads();

  // masked softmax per row; probs f16 overwrite the row's own storage
  const float scale = 0.0510310363079829f;  // 1/sqrt(H) (reference quirk)
  for (int l = tid; l < LP_; l += 256) {
    float mx = -3.0e38f;
    for (int m = 0; m < LP_; ++m) {
      bool valid = (m < L_) && (x[b * L_ + m] != 0);
      float val = valid ? ls[l][m] * scale : -1.0e10f;
      mx = fmaxf(mx, val);
    }
    _Float16* lp = (_Float16*)&ls[l][0];
    float sum = 0.f;
    for (int m = 0; m < LP_; ++m) {
      bool valid = (m < L_) && (x[b * L_ + m] != 0);
      float val = valid ? ls[l][m] * scale : -1.0e10f;  // read before overlay write
      float e = __expf(val - mx);
      sum += e;
      lp[m] = (_Float16)e;
    }
    float rs = 1.f / sum;
    for (int m = 0; m < LP_; ++m) lp[m] = (_Float16)((float)lp[m] * rs);
    for (int m = LP_; m < LK_; ++m) lp[m] = (_Float16)0.f;
  }
  __syncthreads();

  // O = P @ V : 7x2 tiles, K=128 -> 4 wmma each
  for (int t = wave; t < 14; t += 8) {
    int tm = t >> 1, tn = t & 1;
    v8f c = zero;
#pragma unroll
    for (int k0 = 0; k0 < LK_; k0 += 32) {
      V16U af, bf;
      const _Float16* pa = ((const _Float16*)&ls[tm * 16 + mr][0]) + k0 + kbA;
      af.f[0] = *(const f32x4*)pa; af.f[1] = *(const f32x4*)(pa + 16);
      const _Float16* pb = &lvT[tn * 16 + mr][k0 + kbA];
      bf.f[0] = *(const f32x4*)pb; bf.f[1] = *(const f32x4*)(pb + 16);
      c = wmma16(af.v, bf.v, c);
    }
#pragma unroll
    for (int r = 0; r < 8; ++r) {
      int l = tm * 16 + mo + r;
      if (l < L_) {
        int d = tn * 16 + (lane & 15);
        aout[((size_t)b * L_ + l) * H_ + hn * DH_ + d] = (_Float16)c[r];
      }
    }
  }
}

// ---------------------------------------------------------------- residual+LN
// y = cin + resid; z = (g*(y-mu)/sqrt(var_unbiased+1e-8)+be)*nonpad
// optional: acc += z (cumulative-sum skip), f16 mirrors kept in sync.
__global__ __launch_bounds__(128) void resid_ln_kernel(
    const float* __restrict__ cin, const float* __restrict__ resid,
    const int* __restrict__ x, const float* __restrict__ gptr,
    const float* __restrict__ bptr, float* __restrict__ outf,
    _Float16* __restrict__ outh, float* __restrict__ accf,
    _Float16* __restrict__ acch, int update_acc) {
  const int row = blockIdx.x;
  const int tid = threadIdx.x;
  __shared__ float s1[128], s2[128];
  float y[3], sum = 0.f, sq = 0.f;
#pragma unroll
  for (int i = 0; i < 3; ++i) {
    size_t idx = (size_t)row * H_ + tid + i * 128;
    float t = cin[idx] + resid[idx];
    y[i] = t; sum += t; sq += t * t;
  }
  s1[tid] = sum; s2[tid] = sq;
  __syncthreads();
  for (int s = 64; s > 0; s >>= 1) {
    if (tid < s) { s1[tid] += s1[tid + s]; s2[tid] += s2[tid + s]; }
    __syncthreads();
  }
  float mu = s1[0] / (float)H_;
  float var = fmaxf((s2[0] - (float)H_ * mu * mu) / (float)(H_ - 1), 0.f);
  float inv = rsqrtf(var + 1e-8f);
  float g = *gptr, be = *bptr;
  float np = (x[row] != 0) ? 1.f : 0.f;
#pragma unroll
  for (int i = 0; i < 3; ++i) {
    size_t idx = (size_t)row * H_ + tid + i * 128;
    float z = (g * ((y[i] - mu) * inv) + be) * np;
    outf[idx] = z;
    outh[idx] = (_Float16)z;
    if (update_acc) {
      float a = accf[idx] + z;
      accf[idx] = a;
      acch[idx] = (_Float16)a;
    }
  }
}

// ---------------------------------------------------------------- GELU (tanh)
__global__ __launch_bounds__(256) void gelu_cast_kernel(
    const float* __restrict__ c, _Float16* __restrict__ o, size_t n) {
  size_t stride = (size_t)gridDim.x * blockDim.x;
  for (size_t i = (size_t)blockIdx.x * blockDim.x + threadIdx.x; i < n; i += stride) {
    float xv = c[i];
    float t = 0.7978845608028654f * (xv + 0.044715f * xv * xv * xv);
    o[i] = (_Float16)(xv * 0.5f * (1.f + tanhf(t)));
  }
}

// ---------------------------------------------------------------- launcher
extern "C" void kernel_launch(void* const* d_in, const int* in_sizes, int n_in,
                              void* d_out, int out_size, void* d_ws, size_t ws_size,
                              hipStream_t stream) {
  (void)in_sizes; (void)n_in; (void)out_size; (void)ws_size;
  const int*   x   = (const int*)d_in[0];
  const int*   seg = (const int*)d_in[1];
  const float* tok = (const float*)d_in[2];
  const float* se  = (const float*)d_in[3];
  const float* pe  = (const float*)d_in[4];
  const float* Wq  = (const float*)d_in[5];
  const float* bq  = (const float*)d_in[6];
  const float* Wk  = (const float*)d_in[7];
  const float* bk  = (const float*)d_in[8];
  const float* Wv  = (const float*)d_in[9];
  const float* bv  = (const float*)d_in[10];
  const float* Wo  = (const float*)d_in[11];
  const float* bo  = (const float*)d_in[12];
  const float* g1  = (const float*)d_in[13];
  const float* be1 = (const float*)d_in[14];
  const float* W1  = (const float*)d_in[15];
  const float* b1  = (const float*)d_in[16];
  const float* W2  = (const float*)d_in[17];
  const float* b2  = (const float*)d_in[18];
  const float* g2  = (const float*)d_in[19];
  const float* be2 = (const float*)d_in[20];

  char* p = (char*)d_ws;
  auto carve = [&](size_t bytes) -> char* {
    char* r = p;
    p += (bytes + 255) & ~(size_t)255;
    return r;
  };
  const size_t MH = (size_t)M_ * H_;
  const size_t MF = (size_t)M_ * FF_;
  float*    acc   = (float*)carve(MH * 4);
  _Float16* acch  = (_Float16*)carve(MH * 2);
  float*    attnf = (float*)carve(MH * 4);
  _Float16* attnh = (_Float16*)carve(MH * 2);
  float*    qb    = (float*)carve(MH * 4);
  float*    kbuf  = (float*)carve(MH * 4);
  float*    vbuf  = (float*)carve(MH * 4);
  _Float16* ah    = (_Float16*)carve(MH * 2);
  _Float16* outh  = (_Float16*)carve(MH * 2);
  float*    cbuf  = (float*)carve(MF * 4);
  _Float16* f1h   = (_Float16*)carve(MF * 2);
  const size_t WSQ = (size_t)NL_ * H_ * H_;
  const size_t WFF = (size_t)NL_ * H_ * FF_;
  _Float16* wqh = (_Float16*)carve(WSQ * 2);
  _Float16* wkh = (_Float16*)carve(WSQ * 2);
  _Float16* wvh = (_Float16*)carve(WSQ * 2);
  _Float16* woh = (_Float16*)carve(WSQ * 2);
  _Float16* w1h = (_Float16*)carve(WFF * 2);
  _Float16* w2h = (_Float16*)carve(WFF * 2);

  cast_kernel<<<2048, 256, 0, stream>>>(Wq, wqh, WSQ);
  cast_kernel<<<2048, 256, 0, stream>>>(Wk, wkh, WSQ);
  cast_kernel<<<2048, 256, 0, stream>>>(Wv, wvh, WSQ);
  cast_kernel<<<2048, 256, 0, stream>>>(Wo, woh, WSQ);
  cast_kernel<<<4096, 256, 0, stream>>>(W1, w1h, WFF);
  cast_kernel<<<4096, 256, 0, stream>>>(W2, w2h, WFF);

  embed_kernel<<<M_, 128, 0, stream>>>(x, seg, tok, se, pe, acc, acch);

  const dim3 gH(H_ / 128, M_ / 128);    // N=384
  const dim3 gF(FF_ / 128, M_ / 128);   // N=1536

  for (int i = 0; i < NL_; ++i) {
    const _Float16* wq_i = wqh + (size_t)i * H_ * H_;
    const _Float16* wk_i = wkh + (size_t)i * H_ * H_;
    const _Float16* wv_i = wvh + (size_t)i * H_ * H_;
    const _Float16* wo_i = woh + (size_t)i * H_ * H_;
    const _Float16* w1_i = w1h + (size_t)i * H_ * FF_;
    const _Float16* w2_i = w2h + (size_t)i * FF_ * H_;

    gemm_f16<<<gH, 256, 0, stream>>>(acch, wq_i, bq + (size_t)i * H_, qb, H_, H_);
    gemm_f16<<<gH, 256, 0, stream>>>(acch, wk_i, bk + (size_t)i * H_, kbuf, H_, H_);
    gemm_f16<<<gH, 256, 0, stream>>>(acch, wv_i, bv + (size_t)i * H_, vbuf, H_, H_);

    attn_kernel<<<dim3(B_, NH_), 256, 0, stream>>>(qb, kbuf, vbuf, x, ah);

    gemm_f16<<<gH, 256, 0, stream>>>(ah, wo_i, bo + (size_t)i * H_, cbuf, H_, H_);
    resid_ln_kernel<<<M_, 128, 0, stream>>>(cbuf, acc, x, g1 + i, be1 + i,
                                            attnf, attnh, nullptr, nullptr, 0);

    gemm_f16<<<gF, 256, 0, stream>>>(attnh, w1_i, b1 + (size_t)i * FF_, cbuf, FF_, H_);
    gelu_cast_kernel<<<4096, 256, 0, stream>>>(cbuf, f1h, MF);
    gemm_f16<<<gH, 256, 0, stream>>>(f1h, w2_i, b2 + (size_t)i * H_, cbuf, H_, FF_);
    // writes `out` of this layer into d_out (last layer's survives) and acc += out
    resid_ln_kernel<<<M_, 128, 0, stream>>>(cbuf, attnf, x, g2 + i, be2 + i,
                                            (float*)d_out, outh, acc, acch, 1);
  }
}
